// RetNetLayer_67370857005591
// MI455X (gfx1250) — matrix operand order repaired and checked by reading, not compiled, not run on hardware
//
#include <hip/hip_runtime.h>
#include <hip/hip_bf16.h>
#include <math.h>

// ---------------- problem constants ----------------
#define DIM   240
#define SEQ   1024
#define BATCH 240
#define MROWS (BATCH * SEQ)       // 245760 rows of the flattened [b*s, d] matrices
#define KPAD  256                 // 240 padded to 8 * 32 for bf16 WMMA K-steps
#define NT    15                  // 240 / 16 N-tiles
#define KB    8                   // 256 / 32 K-steps
#define WPB   4                   // waves per block (blockDim = 128)

typedef __bf16 bf16;
typedef __attribute__((ext_vector_type(16))) __bf16 v16bf;
typedef __attribute__((ext_vector_type(8)))  __bf16 v8bf;
typedef __attribute__((ext_vector_type(8)))  float  v8f;

// ---------------- fragment helpers (per CDNA5 ISA 7.12.2 layouts) ----------------
// A-matrix 16x32 bf16: lane (l&15)=row; lanes 0-15 hold K={kb..kb+7, kb+16..kb+23},
// lanes 16-31 hold K={kb+8..kb+15, kb+24..kb+31}.
__device__ __forceinline__ v16bf load_a_frag(const bf16* __restrict__ base, int m0,
                                             int kbase, int lane) {
    const int half = lane >> 4;
    const int m    = m0 + (lane & 15);
    const bf16* p  = base + (size_t)m * KPAD + kbase + 8 * half;
    v8bf lo = *(const v8bf*)(p);
    v8bf hi = *(const v8bf*)(p + 16);
    v16bf a;
#pragma unroll
    for (int i = 0; i < 8; ++i) { a[i] = lo[i]; a[i + 8] = hi[i]; }
    return a;
}

// B-matrix 32x16 bf16: lane (l&15)=col n; lanes 0-15 hold K={kb..kb+7, kb+16..kb+23},
// lanes 16-31 hold K={kb+8..kb+15, kb+24..kb+31}. Weight stored as Wb[n][k] (= W[n][k], K padded).
__device__ __forceinline__ v16bf load_b_frag(const bf16* __restrict__ wb, int n0,
                                             int kbase, int lane) {
    const int half = lane >> 4;
    const int n    = n0 + (lane & 15);
    const bf16* p  = wb + (size_t)n * KPAD + kbase + 8 * half;
    v8bf lo = *(const v8bf*)(p);
    v8bf hi = *(const v8bf*)(p + 16);
    v16bf b;
#pragma unroll
    for (int i = 0; i < 8; ++i) { b[i] = lo[i]; b[i + 8] = hi[i]; }
    return b;
}

__device__ __forceinline__ v8f wmma_bf16(v16bf a, v16bf b, v8f c) {
    return __builtin_amdgcn_wmma_f32_16x16x32_bf16(false, a, false, b, (short)0, c,
                                                   false, false);
}

// ---------------- kernel 1: convert x fp32 [M,240] -> bf16 [M,256] (zero-padded) ----
__global__ void k_conv_x(const float* __restrict__ x, bf16* __restrict__ xb) {
    const int t  = blockIdx.x * blockDim.x + threadIdx.x;   // MROWS*32 threads
    const int m  = t >> 5;
    const int kg = (t & 31) * 8;
    v8bf o;
#pragma unroll
    for (int i = 0; i < 8; ++i) {
        const int k = kg + i;
        const float v = (k < DIM) ? x[(size_t)m * DIM + k] : 0.0f;
        o[i] = (bf16)v;
    }
    *(v8bf*)(xb + (size_t)m * KPAD + kg) = o;
}

// ---------------- kernel 2: convert 5 weights [240,240] -> bf16 [240,256] ----------
__global__ void k_conv_w(const float* __restrict__ W0, const float* __restrict__ W1,
                         const float* __restrict__ W2, const float* __restrict__ W3,
                         const float* __restrict__ W4, bf16* __restrict__ dst) {
    const int t = blockIdx.x * blockDim.x + threadIdx.x;    // 5*240*32 threads
    if (t >= 5 * DIM * 32) return;
    const int j  = t / (DIM * 32);
    const int r  = t % (DIM * 32);
    const int n  = r >> 5;
    const int kg = (r & 31) * 8;
    const float* W = (j == 0) ? W0 : (j == 1) ? W1 : (j == 2) ? W2 : (j == 3) ? W3 : W4;
    v8bf o;
#pragma unroll
    for (int i = 0; i < 8; ++i) {
        const int k = kg + i;
        o[i] = (bf16)((k < DIM) ? W[(size_t)n * DIM + k] : 0.0f);
    }
    *(v8bf*)(dst + (size_t)j * (DIM * KPAD) + (size_t)n * KPAD + kg) = o;
}

// ---------------- kernel 3: V = x @ Wv^T (bf16 WMMA, fp32 out) --------------------
__global__ __launch_bounds__(32 * WPB) void k_gemm_v(const bf16* __restrict__ xb,
                                                     const bf16* __restrict__ wvb,
                                                     float* __restrict__ V) {
    const int lane = threadIdx.x & 31;
    const int wave = threadIdx.x >> 5;
    const int m0   = (blockIdx.x * WPB + wave) * 16;
    const int half = lane >> 4;
    const int l15  = lane & 15;

    v8f zero = {};
    v8f acc[NT];
#pragma unroll
    for (int nt = 0; nt < NT; ++nt) acc[nt] = zero;

    if (m0 + 16 < MROWS)  // hint next A tile into cache
        __builtin_prefetch(xb + (size_t)(m0 + 16) * KPAD, 0, 0);

#pragma unroll
    for (int kb = 0; kb < KB; ++kb) {
        const v16bf a = load_a_frag(xb, m0, kb * 32, lane);
#pragma unroll
        for (int nt = 0; nt < NT; ++nt) {
            const v16bf b = load_b_frag(wvb, nt * 16, kb * 32, lane);
            acc[nt] = wmma_bf16(a, b, acc[nt]);
        }
    }
#pragma unroll
    for (int nt = 0; nt < NT; ++nt) {
        const int n = nt * 16 + l15;
#pragma unroll
        for (int r = 0; r < 8; ++r) {
            const int m = m0 + r + 8 * half;
            V[(size_t)m * DIM + n] = acc[nt][r];
        }
    }
}

// ---------------- kernel 4: in-place prefix sum of V over s ------------------------
__global__ void k_scan(float* __restrict__ V) {
    const int t = blockIdx.x * blockDim.x + threadIdx.x;    // BATCH*DIM threads
    if (t >= BATCH * DIM) return;
    const int b = t / DIM, d = t % DIM;
    const size_t base = (size_t)b * SEQ * DIM + d;
    float run = 0.0f;
    for (int s = 0; s < SEQ; ++s) {
        run += V[base + (size_t)s * DIM];
        V[base + (size_t)s * DIM] = run;
    }
}

// ---------------- kernel 5: K = x @ Wk^T fused with diag = decay * <K, Vprefix> ----
__global__ __launch_bounds__(32 * WPB) void k_diag(const bf16* __restrict__ xb,
                                                   const bf16* __restrict__ wkb,
                                                   const float* __restrict__ Vp,
                                                   const float* __restrict__ dm,
                                                   float* __restrict__ diag) {
    const int lane = threadIdx.x & 31;
    const int wave = threadIdx.x >> 5;
    const int m0   = (blockIdx.x * WPB + wave) * 16;
    const int half = lane >> 4;
    const int l15  = lane & 15;

    v8f zero = {};
    v8f acc[NT];
#pragma unroll
    for (int nt = 0; nt < NT; ++nt) acc[nt] = zero;

#pragma unroll
    for (int kb = 0; kb < KB; ++kb) {
        const v16bf a = load_a_frag(xb, m0, kb * 32, lane);
#pragma unroll
        for (int nt = 0; nt < NT; ++nt) {
            const v16bf b = load_b_frag(wkb, nt * 16, kb * 32, lane);
            acc[nt] = wmma_bf16(a, b, acc[nt]);
        }
    }

    float part[8];
#pragma unroll
    for (int r = 0; r < 8; ++r) part[r] = 0.0f;
#pragma unroll
    for (int nt = 0; nt < NT; ++nt) {
        const int n = nt * 16 + l15;
#pragma unroll
        for (int r = 0; r < 8; ++r) {
            const int m = m0 + r + 8 * half;
            part[r] += acc[nt][r] * Vp[(size_t)m * DIM + n];
        }
    }
    // reduce over the 16 n-lanes of each half-wave
#pragma unroll
    for (int mask = 1; mask < 16; mask <<= 1) {
#pragma unroll
        for (int r = 0; r < 8; ++r) part[r] += __shfl_xor(part[r], mask, 32);
    }
    if (l15 == 0) {
        const float decay = dm[0];   // tril fill value (0.9)
#pragma unroll
        for (int r = 0; r < 8; ++r) diag[m0 + r + 8 * half] = decay * part[r];
    }
}

// ------- kernel 6: x_k & g GEMMs -> y = x_k * diag[d,s] * sigmoid(g) -> y@Wo^T+bo --
__global__ __launch_bounds__(32 * WPB) void k_out(const bf16* __restrict__ xb,
                                                  const bf16* __restrict__ wxb,
                                                  const bf16* __restrict__ wgb,
                                                  const bf16* __restrict__ wob,
                                                  const float* __restrict__ diag,
                                                  const float* __restrict__ bo,
                                                  float* __restrict__ out) {
    __shared__ __align__(64) bf16 yt[WPB][16 * KPAD];   // per-wave y tile, A-layout
    const int lane = threadIdx.x & 31;
    const int wave = threadIdx.x >> 5;
    const int m0   = (blockIdx.x * WPB + wave) * 16;
    const int half = lane >> 4;
    const int l15  = lane & 15;
    bf16* my = &yt[wave][0];

    // zero K padding cols 240..255 of the y tile
    {
        const int row = lane >> 1;
        const int c0  = DIM + (lane & 1) * 8;
#pragma unroll
        for (int i = 0; i < 8; ++i) my[row * KPAD + c0 + i] = (bf16)0.0f;
    }

    v8f zero = {};
#pragma unroll
    for (int nt = 0; nt < NT; ++nt) {
        v8f ax = zero, ag = zero;
#pragma unroll
        for (int kb = 0; kb < KB; ++kb) {
            const v16bf a  = load_a_frag(xb, m0, kb * 32, lane);
            const v16bf bx = load_b_frag(wxb, nt * 16, kb * 32, lane);
            const v16bf bg = load_b_frag(wgb, nt * 16, kb * 32, lane);
            ax = wmma_bf16(a, bx, ax);
            ag = wmma_bf16(a, bg, ag);
        }
        const int n = nt * 16 + l15;       // feature index d
#pragma unroll
        for (int r = 0; r < 8; ++r) {
            const int m = m0 + r + 8 * half;
            const int s = m & (SEQ - 1);
            const float dval = diag[(size_t)n * SEQ + s];   // diag[d, s] broadcast
            const float g    = 1.0f / (1.0f + __expf(-ag[r]));
            const float y    = ax[r] * dval * g;
            my[(r + 8 * half) * KPAD + n] = (bf16)y;
        }
    }
    __syncthreads();   // y tile complete (wave-local, but keeps DS ordering simple)

#pragma unroll
    for (int nt2 = 0; nt2 < NT; ++nt2) {
        v8f acc = zero;
#pragma unroll
        for (int kb = 0; kb < KB; ++kb) {
            const v16bf a = load_a_frag(my, 0, kb * 32, lane);
            const v16bf b = load_b_frag(wob, nt2 * 16, kb * 32, lane);
            acc = wmma_bf16(a, b, acc);
        }
        const int n2   = nt2 * 16 + l15;
        const float bb = bo[n2];
#pragma unroll
        for (int r = 0; r < 8; ++r) {
            const int m = m0 + r + 8 * half;
            out[(size_t)m * DIM + n2] = acc[r] + bb;
        }
    }
}

// ---------------- launcher ----------------
extern "C" void kernel_launch(void* const* d_in, const int* in_sizes, int n_in,
                              void* d_out, int out_size, void* d_ws, size_t ws_size,
                              hipStream_t stream) {
    const float* x  = (const float*)d_in[0];
    const float* Wx = (const float*)d_in[1];
    const float* Wk = (const float*)d_in[2];
    const float* Wv = (const float*)d_in[3];
    const float* Wg = (const float*)d_in[4];
    const float* Wo = (const float*)d_in[5];
    const float* bo = (const float*)d_in[6];
    const float* dm = (const float*)d_in[7];
    float* out = (float*)d_out;

    // workspace layout
    const size_t XB_BYTES = (size_t)MROWS * KPAD * sizeof(bf16);   // 125,829,120
    const size_t WB_ELEMS = (size_t)DIM * KPAD;                    // 61,440
    const size_t WB_BYTES = WB_ELEMS * sizeof(bf16);               // 122,880
    char* ws = (char*)d_ws;
    bf16*  xb   = (bf16*)ws;
    bf16*  wall = (bf16*)(ws + XB_BYTES);
    bf16*  wxb  = wall + 0 * WB_ELEMS;
    bf16*  wkb  = wall + 1 * WB_ELEMS;
    bf16*  wvb  = wall + 2 * WB_ELEMS;
    bf16*  wgb  = wall + 3 * WB_ELEMS;
    bf16*  wob  = wall + 4 * WB_ELEMS;
    float* V    = (float*)(ws + XB_BYTES + 5 * WB_BYTES);
    float* diag = (float*)(ws + XB_BYTES + 5 * WB_BYTES + (size_t)MROWS * DIM * sizeof(float));

    // 1) conversions
    k_conv_x<<<(MROWS * 32) / 256, 256, 0, stream>>>(x, xb);
    k_conv_w<<<(5 * DIM * 32 + 255) / 256, 256, 0, stream>>>(Wx, Wk, Wv, Wg, Wo, wall);
    // 2) V = x @ Wv^T
    k_gemm_v<<<MROWS / 16 / WPB, 32 * WPB, 0, stream>>>(xb, wvb, V);
    // 3) prefix scan over s
    k_scan<<<(BATCH * DIM + 255) / 256, 256, 0, stream>>>(V);
    // 4) diag = decay * <x@Wk^T, Vprefix>
    k_diag<<<MROWS / 16 / WPB, 32 * WPB, 0, stream>>>(xb, wkb, V, dm, diag);
    // 5) fused projections + gating + output GEMM
    k_out<<<MROWS / 16 / WPB, 32 * WPB, 0, stream>>>(xb, wxb, wgb, wob, diag, bo, out);
}
